// ContentAM_71468255805546
// MI455X (gfx1250) — matrix operand order repaired and checked by reading, not compiled, not run on hardware
//
#include <hip/hip_runtime.h>
#include <hip/hip_bf16.h>

// Problem constants from the reference
#define CC 1024
#define DD 128
#define RR 64
#define NB 4
#define NN 4096   // H*W = 64*64

typedef __bf16 bf16_t;
typedef bf16_t v16bf __attribute__((ext_vector_type(16)));
typedef float  v8f   __attribute__((ext_vector_type(8)));
typedef unsigned int v4u __attribute__((ext_vector_type(4)));
typedef int v4i __attribute__((ext_vector_type(4)));
typedef int v8i __attribute__((ext_vector_type(8)));

#if defined(__has_builtin)
#if __has_builtin(__builtin_amdgcn_tensor_load_to_lds) && __has_builtin(__builtin_amdgcn_s_wait_tensorcnt)
#define HAVE_TDM 1
#endif
#endif

union Frag { uint4 u[2]; v16bf v; };

__device__ __forceinline__ v16bf load_frag(const bf16_t* p0, const bf16_t* p1) {
  Frag f;
  f.u[0] = *reinterpret_cast<const uint4*>(p0);
  f.u[1] = *reinterpret_cast<const uint4*>(p1);
  return f.v;
}

// A fragment: A stored row-major [M][K] (ld elems/row). 16x32 tile at (m0,k0).
__device__ __forceinline__ v16bf load_a(const bf16_t* A, int ld, int m0, int k0, int lane) {
  const bf16_t* p = A + (size_t)(m0 + (lane & 15)) * ld + k0 + (lane >> 4) * 16;
  return load_frag(p, p + 8);
}

// B fragment: operand stored K-contiguous per column: [N][K] (ld elems/col).
__device__ __forceinline__ v16bf load_b(const bf16_t* Bm, int ld, int n0, int k0, int lane) {
  const bf16_t* p = Bm + (size_t)(n0 + (lane & 15)) * ld + k0 + (lane >> 4) * 16;
  return load_frag(p, p + 8);
}

__device__ __forceinline__ v8f wmma_bf16(v16bf a, v16bf b, v8f c) {
  return __builtin_amdgcn_wmma_f32_16x16x32_bf16(false, a, false, b, (short)0, c, false, false);
}

__device__ __forceinline__ unsigned pack2(float a, float b) {
  union { bf16_t h[2]; unsigned u; } t;
  t.h[0] = (bf16_t)a; t.h[1] = (bf16_t)b;
  return t.u;
}

// ---------------- elementwise prep ----------------

__global__ void k_convert(const float* __restrict__ in, bf16_t* __restrict__ out, int n) {
  int i = blockIdx.x * blockDim.x + threadIdx.x;
  if (i < n) out[i] = (bf16_t)in[i];
}

// x[b][c][n] (fp32) -> xt[b][n][c] (bf16)
__global__ void k_transpose_bf16(const float* __restrict__ x, bf16_t* __restrict__ xt) {
  int i = blockIdx.x * blockDim.x + threadIdx.x;
  if (i >= NB * CC * NN) return;
  int b = i >> 22;
  int rem = i & ((1 << 22) - 1);
  int c = rem >> 12;
  int n = rem & (NN - 1);
  xt[(size_t)b * NN * CC + (size_t)n * CC + c] = (bf16_t)x[i];
}

// ---------------- 1x1-conv projections as WMMA GEMM ----------------
__global__ void k_proj(const bf16_t* __restrict__ W, const float* __restrict__ bias,
                       const bf16_t* __restrict__ Xt, bf16_t* __restrict__ out,
                       int O, int mode) {
  int lane = threadIdx.x & 31;
  int wave = blockIdx.x * (blockDim.x >> 5) + (threadIdx.x >> 5);
  int oT = O >> 4, nT = NN >> 6;
  int total = NB * oT * nT;
  if (wave >= total) return;
  int b = wave / (oT * nT);
  int rem = wave % (oT * nT);
  int ot = rem / nT, nt = rem % nT;
  const bf16_t* Xb = Xt + (size_t)b * NN * CC;
  int m0 = ot * 16, n0 = nt * 64;

  v8f acc[4] = {};
  for (int k = 0; k < CC; k += 32) {
    v16bf a = load_a(W, CC, m0, k, lane);
    v16bf bb0 = load_b(Xb, CC, n0,      k, lane);
    v16bf bb1 = load_b(Xb, CC, n0 + 16, k, lane);
    v16bf bb2 = load_b(Xb, CC, n0 + 32, k, lane);
    v16bf bb3 = load_b(Xb, CC, n0 + 48, k, lane);
    acc[0] = wmma_bf16(a, bb0, acc[0]);
    acc[1] = wmma_bf16(a, bb1, acc[1]);
    acc[2] = wmma_bf16(a, bb2, acc[2]);
    acc[3] = wmma_bf16(a, bb3, acc[3]);
  }

  int r0 = (lane >> 4) * 8;
  float bvv[8];
#pragma unroll
  for (int r = 0; r < 8; ++r) bvv[r] = bias[m0 + r0 + r];
#pragma unroll
  for (int j = 0; j < 4; ++j)
#pragma unroll
    for (int r = 0; r < 8; ++r) acc[j][r] += bvv[r];

  if (mode == 0) {
#pragma unroll
    for (int j = 0; j < 4; ++j) {
      int n = n0 + j * 16 + (lane & 15);
      bf16_t* p = out + (size_t)b * NN * O + (size_t)n * O + m0 + r0;
      uint4 pk;
      pk.x = pack2(acc[j][0], acc[j][1]);
      pk.y = pack2(acc[j][2], acc[j][3]);
      pk.z = pack2(acc[j][4], acc[j][5]);
      pk.w = pack2(acc[j][6], acc[j][7]);
      *reinterpret_cast<uint4*>(p) = pk;
    }
  } else {
#pragma unroll
    for (int j = 0; j < 4; ++j) {
      int n = n0 + j * 16 + (lane & 15);
#pragma unroll
      for (int r = 0; r < 8; ++r)
        out[(size_t)b * O * NN + (size_t)(m0 + r0 + r) * NN + n] = (bf16_t)acc[j][r];
    }
  }
}

// ---------------- fused sigmoid-attention with TDM K-staging ----------------
#define MCHUNK 256
#define MPAD   (MCHUNK + 8)        // att rows: 528B, 16B aligned
#define KROWB  272                 // padded K row: 256B data + 16B TDM pad
#define KLDSB  (MCHUNK * KROWB)    // 69632 B staged K chunk
#define ATTOFF KLDSB
#define SMEMB  (KLDSB + 16 * MPAD * 2)  // 78080 B dynamic LDS

#if defined(HAVE_TDM)
// 1-D TDM copy of one contiguous 64KB K chunk into LDS with row padding:
// data_size=8B, pad 4 dwords (16B) after every 64 dwords (256B row).
__device__ __forceinline__ void tdm_load_k_chunk(const bf16_t* gsrc, unsigned lds_off) {
  unsigned long long ga = (unsigned long long)(const void*)gsrc;
  v4u g0;
  g0.x = 1u;                                    // count=1, user descriptor
  g0.y = lds_off;                               // lds_addr (bytes)
  g0.z = (unsigned)ga;                          // global_addr[31:0]
  g0.w = ((unsigned)(ga >> 32) & 0x1FFFFFFu) | (2u << 30);  // addr[56:32], type=2
  const unsigned tensor_dim0 = 1u << 22;        // large: no OOB clipping
  const unsigned tile_dim0 = (MCHUNK * DD * 2) / 8;  // 8192 x 8B units
  v8i g1;
  g1[0] = (int)((3u << 16)                      // data_size = 8B
              | (1u << 20)                      // pad_enable
              | (5u << 22)                      // pad_interval: 64 dwords
              | (3u << 25));                    // pad_amount: 4 dwords
  g1[1] = (int)((tensor_dim0 & 0xFFFFu) << 16); // tensor_dim0[15:0]
  g1[2] = (int)((tensor_dim0 >> 16) | (1u << 16)); // tensor_dim0[31:16], tensor_dim1=1
  g1[3] = (int)(tile_dim0 << 16);               // tensor_dim1 hi=0, tile_dim0
  g1[4] = 0;                                    // tile_dim1=0, tile_dim2=0 (1-D)
  g1[5] = (int)tile_dim0;                       // tensor_dim0_stride (unused)
  g1[6] = 0;
  g1[7] = 0;
  v4i z4 = {};
#if defined(__clang_major__) && __clang_major__ >= 23
  v8i z8 = {};
  __builtin_amdgcn_tensor_load_to_lds(g0, g1, z4, z4, z8, 0);
#else
  __builtin_amdgcn_tensor_load_to_lds(g0, g1, z4, z4, 0);
#endif
}
#endif

__device__ __forceinline__ void stage_issue(const bf16_t* gsrc, char* smem, int tid) {
#if defined(HAVE_TDM)
  if ((tid >> 5) == 0)
    tdm_load_k_chunk(gsrc, __builtin_amdgcn_groupstaticsize());
#else
  const unsigned long long* s = (const unsigned long long*)gsrc;
  for (int u = tid; u < (MCHUNK * DD * 2) / 8; u += 256) {
    int row = u >> 5;                 // 32 x 8B units per 256B row
    *(unsigned long long*)(smem + row * KROWB + (u & 31) * 8) = s[u];
  }
#endif
}

__device__ __forceinline__ void stage_wait(int tid) {
#if defined(HAVE_TDM)
  if ((tid >> 5) == 0) __builtin_amdgcn_s_wait_tensorcnt(0);
#endif
}

__global__ void k_attn(const bf16_t* __restrict__ Qt, const bf16_t* __restrict__ Kt,
                       const bf16_t* __restrict__ V, float* __restrict__ out) {
  extern __shared__ __align__(16) char smem[];
  bf16_t* ldsK = (bf16_t*)smem;                          // padded [256 m][136] rows
  bf16_t (*attS)[MPAD] = (bf16_t (*)[MPAD])(smem + ATTOFF);  // [16 n][m]

  int tid = threadIdx.x;
  int lane = tid & 31;
  int w = tid >> 5;
  const int cB = CC / 256;           // 4 c-blocks
  const int nT = NN / 16;            // 256 n-tiles
  int blk = blockIdx.x;
  int b = blk / (cB * nT);
  int rem = blk % (cB * nT);
  int cb = rem / nT, nt = rem % nT;

  const bf16_t* Qb = Qt + (size_t)b * NN * DD;
  const bf16_t* Kb = Kt + (size_t)b * NN * DD;
  const bf16_t* Vb = V + (size_t)b * CC * NN;
  int c0 = cb * 256 + w * 32;        // this wave's 32 output rows

  // Q A-fragments are m-invariant: preload all 4 K-steps of D=128.
  v16bf aQ0 = load_a(Qb, DD, nt * 16,  0, lane);
  v16bf aQ1 = load_a(Qb, DD, nt * 16, 32, lane);
  v16bf aQ2 = load_a(Qb, DD, nt * 16, 64, lane);
  v16bf aQ3 = load_a(Qb, DD, nt * 16, 96, lane);

  // Stage first K chunk (TDM DMA global->LDS, row-padded).
  stage_issue(Kb, smem, tid);
  stage_wait(tid);
  __syncthreads();

  v8f o0 = {}, o1 = {};
  for (int M = 0; M < NN; M += MCHUNK) {
    // ---- phase 1: this wave's 32-m slice of sigmoid(Q^T K) from LDS K ----
    int mLoc = w * 32;
#pragma unroll
    for (int ms = 0; ms < 2; ++ms) {
      v8f e = {};
      e = wmma_bf16(aQ0, load_b(ldsK, KROWB / 2, mLoc + ms * 16,  0, lane), e);
      e = wmma_bf16(aQ1, load_b(ldsK, KROWB / 2, mLoc + ms * 16, 32, lane), e);
      e = wmma_bf16(aQ2, load_b(ldsK, KROWB / 2, mLoc + ms * 16, 64, lane), e);
      e = wmma_bf16(aQ3, load_b(ldsK, KROWB / 2, mLoc + ms * 16, 96, lane), e);
#pragma unroll
      for (int r = 0; r < 8; ++r) {
        float s = 1.0f / (1.0f + __expf(-e[r]));
        attS[(lane >> 4) * 8 + r][w * 32 + ms * 16 + (lane & 15)] = (bf16_t)s;
      }
    }
    // Prefetch next chunk of this wave's V rows.
    if (M + MCHUNK < NN) {
      __builtin_prefetch(Vb + (size_t)(c0 + (lane & 15)) * NN + M + MCHUNK, 0, 3);
      __builtin_prefetch(Vb + (size_t)(c0 + 16 + (lane & 15)) * NN + M + MCHUNK, 0, 3);
    }
    __syncthreads();   // attS ready; all LDS-K reads for chunk M complete

    // Pipeline the DMA of chunk M+1 under phase-2 compute.
    if (M + MCHUNK < NN) stage_issue(Kb + (size_t)(M + MCHUNK) * DD, smem, tid);

    // ---- phase 2: out-tile WMMAs over the shared 256-m att chunk ----
#pragma unroll
    for (int km = 0; km < 8; ++km) {
      const bf16_t* pb = &attS[lane & 15][km * 32 + (lane >> 4) * 16];
      v16bf b2 = load_frag(pb, pb + 8);
      v16bf av0 = load_a(Vb, NN, c0,      M + km * 32, lane);
      v16bf av1 = load_a(Vb, NN, c0 + 16, M + km * 32, lane);
      o0 = wmma_bf16(av0, b2, o0);
      o1 = wmma_bf16(av1, b2, o1);
    }
    stage_wait(tid);   // issuing wave blocks until chunk M+1 landed
    __syncthreads();   // attS consumed; LDS K(M+1) visible to all
  }

  // Store fp32 out [b][c][n].
  int n = nt * 16 + (lane & 15);
  int r0 = (lane >> 4) * 8;
#pragma unroll
  for (int r = 0; r < 8; ++r) {
    out[(size_t)b * CC * NN + (size_t)(c0 + r0 + r) * NN + n] = o0[r];
    out[(size_t)b * CC * NN + (size_t)(c0 + 16 + r0 + r) * NN + n] = o1[r];
  }
}

// ---------------- CBAM channel gate ----------------

__global__ void k_reduce(const float* __restrict__ out, float* __restrict__ avg,
                         float* __restrict__ mx) {
  int lane = threadIdx.x & 31;
  int row = blockIdx.x * (blockDim.x >> 5) + (threadIdx.x >> 5);
  if (row >= NB * CC) return;
  const float* p = out + (size_t)row * NN;
  float s = 0.f, m = -3.4e38f;
  for (int i = lane; i < NN; i += 32) {
    float v = p[i];
    s += v;
    m = fmaxf(m, v);
  }
  for (int off = 16; off > 0; off >>= 1) {
    s += __shfl_xor(s, off);
    m = fmaxf(m, __shfl_xor(m, off));
  }
  if (lane == 0) { avg[row] = s * (1.0f / NN); mx[row] = m; }
}

__global__ void k_gate(const float* __restrict__ avg, const float* __restrict__ mx,
                       const float* __restrict__ W1, const float* __restrict__ W2,
                       float* __restrict__ gate) {
  __shared__ float h[RR];
  int b = blockIdx.x, t = threadIdx.x;
  if (t < RR) {
    float sa = 0.f, sm = 0.f;
    const float* wr = W1 + (size_t)t * CC;
    for (int c = 0; c < CC; ++c) {
      sa += wr[c] * avg[b * CC + c];
      sm += wr[c] * mx[b * CC + c];
    }
    h[t] = fmaxf(sa, 0.f) + fmaxf(sm, 0.f);
  }
  __syncthreads();
  for (int c = t; c < CC; c += blockDim.x) {
    float g = 0.f;
    for (int r = 0; r < RR; ++r) g += W2[(size_t)c * RR + r] * h[r];
    gate[b * CC + c] = 1.0f / (1.0f + __expf(-g));
  }
}

__global__ void k_scale(float* __restrict__ out, const float* __restrict__ gate) {
  size_t i = (size_t)blockIdx.x * blockDim.x + threadIdx.x;
  if (i < (size_t)NB * CC * NN) out[i] *= gate[i >> 12];
}

// ---------------- launch ----------------

extern "C" void kernel_launch(void* const* d_in, const int* in_sizes, int n_in,
                              void* d_out, int out_size, void* d_ws, size_t ws_size,
                              hipStream_t stream) {
  (void)in_sizes; (void)n_in; (void)out_size; (void)ws_size;
  const float* x    = (const float*)d_in[0];
  const float* x1   = (const float*)d_in[1];
  const float* Wq   = (const float*)d_in[2];
  const float* bq   = (const float*)d_in[3];
  const float* Wk   = (const float*)d_in[4];
  const float* bk   = (const float*)d_in[5];
  const float* Wv   = (const float*)d_in[6];
  const float* bv   = (const float*)d_in[7];
  const float* Wca1 = (const float*)d_in[8];
  const float* Wca2 = (const float*)d_in[9];
  float* out = (float*)d_out;

  char* ws = (char*)d_ws;
  size_t off = 0;
  auto alloc = [&](size_t bytes) {
    size_t p = off;
    off += (bytes + 255) & ~(size_t)255;
    return p;
  };
  bf16_t* xt   = (bf16_t*)(ws + alloc((size_t)NB * NN * CC * 2));
  bf16_t* x1t  = (bf16_t*)(ws + alloc((size_t)NB * NN * CC * 2));
  bf16_t* wqb  = (bf16_t*)(ws + alloc((size_t)DD * CC * 2));
  bf16_t* wkb  = (bf16_t*)(ws + alloc((size_t)DD * CC * 2));
  bf16_t* wvb  = (bf16_t*)(ws + alloc((size_t)CC * CC * 2));
  bf16_t* Qt   = (bf16_t*)(ws + alloc((size_t)NB * NN * DD * 2));
  bf16_t* Kt   = (bf16_t*)(ws + alloc((size_t)NB * NN * DD * 2));
  bf16_t* Vbuf = (bf16_t*)(ws + alloc((size_t)NB * CC * NN * 2));
  float* avg   = (float*)(ws + alloc((size_t)NB * CC * 4));
  float* mx    = (float*)(ws + alloc((size_t)NB * CC * 4));
  float* gate  = (float*)(ws + alloc((size_t)NB * CC * 4));

  int tot = NB * CC * NN;
  k_transpose_bf16<<<(tot + 255) / 256, 256, 0, stream>>>(x, xt);
  k_transpose_bf16<<<(tot + 255) / 256, 256, 0, stream>>>(x1, x1t);
  k_convert<<<(DD * CC + 255) / 256, 256, 0, stream>>>(Wq, wqb, DD * CC);
  k_convert<<<(DD * CC + 255) / 256, 256, 0, stream>>>(Wk, wkb, DD * CC);
  k_convert<<<(CC * CC + 255) / 256, 256, 0, stream>>>(Wv, wvb, CC * CC);

  {
    int waves = NB * (DD / 16) * (NN / 64);    // 2048
    k_proj<<<waves / 8, 256, 0, stream>>>(wqb, bq, xt, Qt, DD, 0);
    k_proj<<<waves / 8, 256, 0, stream>>>(wkb, bk, xt, Kt, DD, 0);
  }
  {
    int waves = NB * (CC / 16) * (NN / 64);    // 16384
    k_proj<<<waves / 8, 256, 0, stream>>>(wvb, bv, x1t, Vbuf, CC, 1);
  }
  {
    int blocks = NB * (CC / 256) * (NN / 16);  // 4096 cooperative blocks
    k_attn<<<blocks, 256, SMEMB, stream>>>(Qt, Kt, Vbuf, out);
  }
  k_reduce<<<(NB * CC) / 8, 256, 0, stream>>>(out, avg, mx);
  k_gate<<<NB, 256, 0, stream>>>(avg, mx, Wca1, Wca2, gate);
  k_scale<<<(tot + 255) / 256, 256, 0, stream>>>(out, gate);
}
